// T5Attention_20650202759608
// MI455X (gfx1250) — matrix operand order repaired and checked by reading, not compiled, not run on hardware
//
#include <hip/hip_runtime.h>
#include <stdint.h>

typedef _Float16 f16;
typedef __attribute__((ext_vector_type(16))) _Float16 v16h;
typedef __attribute__((ext_vector_type(8)))  _Float16 v8h;
typedef __attribute__((ext_vector_type(8)))  float    v8f;
typedef __attribute__((ext_vector_type(4)))  uint32_t v4u;
typedef __attribute__((ext_vector_type(8)))  int      v8i;
typedef __attribute__((ext_vector_type(4)))  int      v4i;

union AFrag { v16h v; v8h h[2]; };

#define WMMA_F32_F16(a, b, c) \
  __builtin_amdgcn_wmma_f32_16x16x32_f16(false, (a), false, (b), (short)0, (c), false, false)

#if defined(__has_builtin)
#if __has_builtin(__builtin_amdgcn_tensor_load_to_lds) && \
    __has_builtin(__builtin_amdgcn_s_wait_tensorcnt)
#define HAVE_TDM 1
#endif
#endif
#ifndef HAVE_TDM
#define HAVE_TDM 0
#endif

// ---------------------------------------------------------------------------
// f32 -> f16 conversion (grid-stride)
// ---------------------------------------------------------------------------
__global__ void cvt_kernel(const float* __restrict__ src, f16* __restrict__ dst, int n) {
  int i = blockIdx.x * blockDim.x + threadIdx.x;
  int stride = gridDim.x * blockDim.x;
  for (; i < n; i += stride) dst[i] = (f16)src[i];
}

#if HAVE_TDM
// ---------------------------------------------------------------------------
// Issue one TDM 2D tile load: rows x 32 f16 elements from a 4096x4096 f16
// tensor (row stride 4096) into LDS at lds_off, with 32B padding after each
// 64B row so the LDS image has a 48-half row stride (bank-friendly, and
// exactly what the WMMA fragment loads below expect).
// D# layout per CDNA5 ISA ch.8 (async_tensor). 6-arg builtin form
// (clang-23 / therock-10.0): (v4u g0, v8i g1, v4i g2, v4i g3, v8i, i32 cpol).
// ---------------------------------------------------------------------------
__device__ __forceinline__ void tdm_load_tile(const f16* gptr, uint32_t lds_off,
                                              int rows) {
  const uint64_t ga = (uint64_t)(uintptr_t)gptr;
  v4u g0;
  g0[0] = 1u;                                   // count=1 (valid), user mode
  g0[1] = lds_off;                              // lds_addr (bytes)
  g0[2] = (uint32_t)ga;                         // global_addr[31:0]
  g0[3] = (uint32_t)((ga >> 32) & 0x01FFFFFFu)  // global_addr[56:32]
          | (2u << 30);                         // type=2 ("image")
  v8i g1;
  g1[0] = (1 << 16)                             // data_size = 2 bytes
        | (1 << 20)                             // pad_enable
        | (3 << 22)                             // pad_interval: 16 DWORDs (64B row)
        | (7 << 25);                            // pad_amount: 8 DWORDs (32B)
  g1[1] = (int)((4096u & 0xFFFFu) << 16);       // tensor_dim0[15:0] << 16
  g1[2] = (int)((4096u & 0xFFFFu) << 16);       // dim0 hi=0 | tensor_dim1[15:0]<<16
  g1[3] = (32 << 16);                           // dim1 hi=0 | tile_dim0 = 32
  g1[4] = rows;                                 // tile_dim1 = rows, tile_dim2 = 0
  g1[5] = 4096;                                 // tensor_dim0_stride[31:0]
  g1[6] = 0;                                    // stride hi | dim1_stride lo
  g1[7] = 0;
  const v4i z4 = {0, 0, 0, 0};                  // 2D tensor: groups 2/3 unused
  const v8i z8 = {0, 0, 0, 0, 0, 0, 0, 0};
  __builtin_amdgcn_tensor_load_to_lds(g0, g1, z4, z4, z8, 0);
}
#endif

// ---------------------------------------------------------------------------
// GEMM: D[M,N] = A[M,K] * B[N,K]^T, M=N=K=4096, f16 in, f16/f32 out.
// Block tile 128(M) x 64(N), 8 waves (4x2), wave tile 32x32 = 2x2 WMMA tiles.
// Tiles staged into LDS by the Tensor Data Mover (wave 0 issues, TENSORcnt
// waits), fallback to manual vector copies when the builtin is unavailable.
// ---------------------------------------------------------------------------
template <typename OutT>
__global__ __launch_bounds__(256)
void gemm_nt_kernel(const f16* __restrict__ A, const f16* __restrict__ B,
                    OutT* __restrict__ Dm) {
  __shared__ __align__(32) f16 As[128 * 48];
  __shared__ __align__(32) f16 Bs[64 * 48];

  const int tid   = threadIdx.x;
  const int mBase = blockIdx.y * 128;
  const int nBase = blockIdx.x * 64;
  const int wid   = tid >> 5;
  const int lane  = tid & 31;
  const int lm    = lane & 15;
  const int hf    = lane >> 4;          // half-wave select
  const int wm    = (wid & 3) * 32;     // wave M offset in block tile
  const int wn    = (wid >> 2) * 32;    // wave N offset in block tile

  v8f acc[2][2];
#pragma unroll
  for (int i = 0; i < 2; ++i)
#pragma unroll
    for (int j = 0; j < 2; ++j) acc[i][j] = {};

  const int arow = tid >> 1;            // 0..127
  const int aseg = (tid & 1) << 4;      // 0 or 16 halves

#if HAVE_TDM
  const uint32_t ldsA = (uint32_t)(uintptr_t)&As[0];
  const uint32_t ldsB = (uint32_t)(uintptr_t)&Bs[0];
#endif

  for (int kb = 0; kb < 4096; kb += 32) {
#if HAVE_TDM
    if (wid == 0) {
      tdm_load_tile(&A[(size_t)mBase * 4096 + kb], ldsA, 128);
      tdm_load_tile(&B[(size_t)nBase * 4096 + kb], ldsB, 64);
      __builtin_amdgcn_s_wait_tensorcnt(0);
    }
#else
    // Manual staging: A tile 128 rows x 32 halves, B tile 64 rows x 32 halves
    *(v16h*)&As[arow * 48 + aseg] =
        *(const v16h*)&A[(size_t)(mBase + arow) * 4096 + kb + aseg];
    if (tid < 128)
      *(v16h*)&Bs[arow * 48 + aseg] =
          *(const v16h*)&B[(size_t)(nBase + arow) * 4096 + kb + aseg];
#endif
    if (kb + 32 < 4096)  // gfx1250 global_prefetch_b8 for next K block
      __builtin_prefetch(&A[(size_t)(mBase + arow) * 4096 + kb + 32 + aseg], 0, 0);
    __syncthreads();

    AFrag af[2], bf[2];
#pragma unroll
    for (int i = 0; i < 2; ++i) {
      // A 16x32 f16 layout: lane<16 -> K 0..15; lane>=16 -> K 16..31
      const f16* ap = &As[(wm + i * 16 + lm) * 48 + (hf << 4)];
      af[i].h[0] = *(const v8h*)ap;
      af[i].h[1] = *(const v8h*)(ap + 8);
    }
#pragma unroll
    for (int j = 0; j < 2; ++j) {
      // B 32x16 f16 layout: lane<16 -> col n, K 0..15; lane>=16 -> col n, K 16..31
      bf[j].v = *(const v16h*)&Bs[(wn + j * 16 + lm) * 48 + (hf << 4)];
    }
#pragma unroll
    for (int i = 0; i < 2; ++i)
#pragma unroll
      for (int j = 0; j < 2; ++j)
        acc[i][j] = WMMA_F32_F16(af[i].v, bf[j].v, acc[i][j]);
    __syncthreads();
  }

  // C/D layout: VGPR r -> row r + 8*hf, lane%16 -> col
#pragma unroll
  for (int i = 0; i < 2; ++i)
#pragma unroll
    for (int j = 0; j < 2; ++j)
#pragma unroll
      for (int r = 0; r < 8; ++r) {
        const int row = mBase + wm + i * 16 + r + (hf << 3);
        const int col = nBase + wn + j * 16 + lm;
        Dm[(size_t)row * 4096 + col] = (OutT)acc[i][j][r];
      }
}

// ---------------------------------------------------------------------------
// V transpose: Vh[B*L, 4096] -> Vt[B][H][Dh][L] so attention B-fragments are
// contiguous 32B loads.
// ---------------------------------------------------------------------------
__global__ void transpose_v_kernel(const f16* __restrict__ Vh, f16* __restrict__ Vt) {
  const int idx = blockIdx.x * 256 + threadIdx.x;       // 2^24 total
  const int l  = idx & 511;
  const int dh = (idx >> 9) & 63;
  const int h  = (idx >> 15) & 63;
  const int b  = idx >> 21;
  Vt[idx] = Vh[(size_t)(b * 512 + l) * 4096 + h * 64 + dh];
}

// ---------------------------------------------------------------------------
// T5 relative-position bias. bias[h][q][k] = rel_bias[bucket(k - q)][h]
// ---------------------------------------------------------------------------
__device__ __forceinline__ int rel_bucket(int rp) {
  int ret = rp > 0 ? 16 : 0;
  int arp = rp < 0 ? -rp : rp;
  if (arp < 8) return ret + arp;
  // 8 + int(log(|rp|/8)/log(128/8) * 8), clamped to 15
  int large = 8 + (int)(__logf((float)arp * 0.125f) * 2.8853900817779268f);
  large = large < 15 ? large : 15;
  return ret + large;
}

__global__ void bias_kernel(const float* __restrict__ rb, float* __restrict__ out) {
  const int idx  = blockIdx.x * 256 + threadIdx.x;      // 64*512*512 = 2^24
  const int kcol = idx & 511;
  const int q    = (idx >> 9) & 511;
  const int h    = idx >> 18;
  out[idx] = rb[rel_bucket(kcol - q) * 64 + h];
}

// ---------------------------------------------------------------------------
// Flash attention: one wave per (b, h, 16-query tile). scores = q.k^T + bias
// (the reference's k-prescale cancels the 1/sqrt(Dh)). Online softmax with
// half-wave shfl_xor row reductions; P restaged via LDS into A-frag layout.
// ---------------------------------------------------------------------------
__global__ __launch_bounds__(32)
void attn_kernel(const f16* __restrict__ Qh, const f16* __restrict__ Kh,
                 const f16* __restrict__ Vt, const float* __restrict__ bias,
                 f16* __restrict__ Oh) {
  __shared__ __align__(32) f16 Ps[16 * 40];

  const int q0   = blockIdx.x * 16;
  const int h    = blockIdx.y;
  const int b    = blockIdx.z;
  const int lane = threadIdx.x;
  const int lm   = lane & 15;
  const int hf   = lane >> 4;

  // Q fragments (16 rows x 64 Dh = 2 k-steps), kept in registers all loop.
  AFrag qf[2];
#pragma unroll
  for (int st = 0; st < 2; ++st) {
    const f16* qp = &Qh[(size_t)(b * 512 + q0 + lm) * 4096 + h * 64 + st * 32 + (hf << 4)];
    qf[st].h[0] = *(const v8h*)qp;
    qf[st].h[1] = *(const v8h*)(qp + 8);
  }

  v8f o[4];
#pragma unroll
  for (int dt = 0; dt < 4; ++dt) o[dt] = {};
  float m_run[8], l_run[8];
#pragma unroll
  for (int r = 0; r < 8; ++r) { m_run[r] = -3.0e38f; l_run[r] = 0.0f; }

  const float* bh = bias + ((size_t)h << 18);

  for (int kc = 0; kc < 16; ++kc) {
    const int kn = kc * 32;

    v8f s[2]; s[0] = {}; s[1] = {};
#pragma unroll
    for (int nt = 0; nt < 2; ++nt)
#pragma unroll
      for (int st = 0; st < 2; ++st) {
        AFrag kf;  // B 32x16: lane -> key column, contiguous 16 Dh values
        kf.v = *(const v16h*)&Kh[(size_t)(b * 512 + kn + nt * 16 + lm) * 4096 +
                                 h * 64 + st * 32 + (hf << 4)];
        s[nt] = WMMA_F32_F16(qf[st].v, kf.v, s[nt]);
      }

    float p0[8], p1[8];
#pragma unroll
    for (int r = 0; r < 8; ++r) {
      const int row = q0 + r + (hf << 3);
      float s0 = s[0][r] + bh[(size_t)row * 512 + kn + lm];
      float s1 = s[1][r] + bh[(size_t)row * 512 + kn + 16 + lm];
      float mx = fmaxf(s0, s1);
      mx = fmaxf(mx, __shfl_xor(mx, 1, 32));
      mx = fmaxf(mx, __shfl_xor(mx, 2, 32));
      mx = fmaxf(mx, __shfl_xor(mx, 4, 32));
      mx = fmaxf(mx, __shfl_xor(mx, 8, 32));
      const float mnew = fmaxf(m_run[r], mx);
      const float corr = __expf(m_run[r] - mnew);
      m_run[r] = mnew;
      p0[r] = __expf(s0 - mnew);
      p1[r] = __expf(s1 - mnew);
      float prs = p0[r] + p1[r];
      prs += __shfl_xor(prs, 1, 32);
      prs += __shfl_xor(prs, 2, 32);
      prs += __shfl_xor(prs, 4, 32);
      prs += __shfl_xor(prs, 8, 32);
      l_run[r] = l_run[r] * corr + prs;
#pragma unroll
      for (int dt = 0; dt < 4; ++dt) o[dt][r] *= corr;
    }

    __syncthreads();  // previous iteration's Ps reads done
#pragma unroll
    for (int r = 0; r < 8; ++r) {
      Ps[(r + (hf << 3)) * 40 + lm]      = (f16)p0[r];
      Ps[(r + (hf << 3)) * 40 + 16 + lm] = (f16)p1[r];
    }
    __syncthreads();

    AFrag pa;  // reload P in A-frag layout
    const f16* pp = &Ps[lm * 40 + (hf << 4)];
    pa.h[0] = *(const v8h*)pp;
    pa.h[1] = *(const v8h*)(pp + 8);

#pragma unroll
    for (int dt = 0; dt < 4; ++dt) {
      AFrag vf;  // B 32x16: lane -> Dh column, 16 contiguous keys
      vf.v = *(const v16h*)&Vt[(size_t)((b * 64 + h) * 64 + dt * 16 + lm) * 512 +
                               kn + (hf << 4)];
      o[dt] = WMMA_F32_F16(pa.v, vf.v, o[dt]);
    }
  }

#pragma unroll
  for (int r = 0; r < 8; ++r) {
    const float inv = 1.0f / l_run[r];
    const int row = b * 512 + q0 + r + (hf << 3);
#pragma unroll
    for (int dt = 0; dt < 4; ++dt)
      Oh[(size_t)row * 4096 + h * 64 + dt * 16 + lm] = (f16)(o[dt][r] * inv);
  }
}

// ---------------------------------------------------------------------------
// Launch
// ---------------------------------------------------------------------------
extern "C" void kernel_launch(void* const* d_in, const int* in_sizes, int n_in,
                              void* d_out, int out_size, void* d_ws, size_t ws_size,
                              hipStream_t stream) {
  (void)in_sizes; (void)n_in; (void)out_size; (void)ws_size;

  const float* x  = (const float*)d_in[0];
  const float* Wq = (const float*)d_in[1];
  const float* Wk = (const float*)d_in[2];
  const float* Wv = (const float*)d_in[3];
  const float* Wo = (const float*)d_in[4];
  const float* rb = (const float*)d_in[5];

  float* out      = (float*)d_out;                 // [8,512,4096] f32
  float* bias_out = out + (size_t)16777216;        // [1,64,512,512] f32

  const size_t SZ = 16777216;                       // 4096*4096 elements
  f16* Xh  = (f16*)d_ws;
  f16* Wqh = Xh  + SZ;
  f16* Wkh = Wqh + SZ;
  f16* Wvh = Wkh + SZ;
  f16* Woh = Wvh + SZ;
  f16* Qh  = Woh + SZ;
  f16* Kh  = Qh  + SZ;
  f16* Vh  = Kh  + SZ;
  f16* Vt  = Vh  + SZ;
  f16* Oh  = Xh;  // reuse X buffer once projections are done

  cvt_kernel<<<4096, 256, 0, stream>>>(x,  Xh,  (int)SZ);
  cvt_kernel<<<4096, 256, 0, stream>>>(Wq, Wqh, (int)SZ);
  cvt_kernel<<<4096, 256, 0, stream>>>(Wk, Wkh, (int)SZ);
  cvt_kernel<<<4096, 256, 0, stream>>>(Wv, Wvh, (int)SZ);
  cvt_kernel<<<4096, 256, 0, stream>>>(Wo, Woh, (int)SZ);

  dim3 gGrid(64, 32);  // N/64 x M/128
  gemm_nt_kernel<f16><<<gGrid, 256, 0, stream>>>(Xh, Wqh, Qh);
  gemm_nt_kernel<f16><<<gGrid, 256, 0, stream>>>(Xh, Wkh, Kh);
  gemm_nt_kernel<f16><<<gGrid, 256, 0, stream>>>(Xh, Wvh, Vh);

  transpose_v_kernel<<<65536, 256, 0, stream>>>(Vh, Vt);
  bias_kernel<<<65536, 256, 0, stream>>>(rb, bias_out);

  attn_kernel<<<dim3(32, 64, 8), 32, 0, stream>>>(Qh, Kh, Vt, bias_out, Oh);

  gemm_nt_kernel<float><<<gGrid, 256, 0, stream>>>(Oh, Woh, out);
}
